// SentenceToDocment_57698590655216
// MI455X (gfx1250) — matrix operand order repaired and checked by reading, not compiled, not run on hardware
//
#include <hip/hip_runtime.h>

#define B_   128
#define S_   128
#define IN_  512
#define H_   256
#define C_   256
#define G3_  768   // 3*H

typedef __bf16 bf16;
typedef __attribute__((ext_vector_type(16))) __bf16 v16bf;
typedef __attribute__((ext_vector_type(8)))  float  v8f;

__device__ __forceinline__ v8f wmma_bf16(v16bf a, v16bf b, v8f c) {
  // (neg_a, A, neg_b, B, c_mod, C, reuse_a, reuse_b)
  return __builtin_amdgcn_wmma_f32_16x16x32_bf16(false, a, false, b, (short)0, c, false, false);
}

__device__ __forceinline__ float sigmoidf_(float x) { return 1.0f / (1.0f + __expf(-x)); }

// A-tile fragment load: 16x32 bf16, rows from `arow` (per-lane row base), K window k0
__device__ __forceinline__ v16bf load_a(const bf16* __restrict__ arow, int k0, int hi) {
  v16bf a;
#pragma unroll
  for (int i = 0; i < 8; ++i) {
    int kb = (i < 4 ? 2 * i : 2 * i + 8) + 8 * hi + k0;
    a[2 * i]     = arow[kb];
    a[2 * i + 1] = arow[kb + 1];
  }
  return a;
}

// B-tile fragment load: 32x16 bf16 from weight row base (per-lane col row), K window k0
__device__ __forceinline__ v16bf load_b(const bf16* __restrict__ brow, int k0, int hi) {
  const bf16* bp = brow + k0 + 8 * hi;
  v16bf b;
#pragma unroll
  for (int i = 0; i < 16; ++i) b[i] = bp[i < 8 ? i : i + 8];
  return b;
}

// ---------------------------------------------------------------- f32 -> bf16
__global__ void k_cvt_bf16(const float* __restrict__ src, bf16* __restrict__ dst, int n) {
  int i = blockIdx.x * blockDim.x + threadIdx.x;
  int stride = gridDim.x * blockDim.x;
  for (; i < n; i += stride) dst[i] = (bf16)src[i];
}

// ------------------------------------------------- input projections (both dirs)
// xp[dir][t][b][g] = sum_i x[b, t_eff, i] * w_ih_dir[g, i] + b_ih_dir[g]
// Each wave: one 16-row b tile x FOUR 16-col g tiles; all operand fragments are
// fetched up-front per k-step so loads batch into one clause and overlap WMMAs.
__global__ void k_input_proj(const bf16* __restrict__ xbf,   // [B][S][IN]
                             const bf16* __restrict__ wih,   // [2][G3*IN]
                             const float* __restrict__ bih_f,
                             const float* __restrict__ bih_b,
                             float* __restrict__ xp)          // [2][S][B][G3]
{
  const int nt   = blockIdx.x;            // 0..11 : 64-wide column block of G3
  const int t    = blockIdx.y;            // 0..127
  const int dir  = blockIdx.z;            // 0..1
  const int wave = threadIdx.x >> 5;      // 0..7 : 16-row b sub-tile
  const int lane = threadIdx.x & 31;
  const int lm = lane & 15, hi = lane >> 4;

  const int t_eff = dir ? (S_ - 1 - t) : t;
  const int n0 = nt * 64;
  const int b0 = wave * 16;
  const bf16*  W   = wih + (size_t)dir * G3_ * IN_;
  const float* bih = dir ? bih_b : bih_f;

  const bf16* arow = xbf + (((size_t)(b0 + lm) * S_) + t_eff) * IN_;
  const bf16* brow0 = W + (size_t)(n0 +  0 + lm) * IN_;
  const bf16* brow1 = W + (size_t)(n0 + 16 + lm) * IN_;
  const bf16* brow2 = W + (size_t)(n0 + 32 + lm) * IN_;
  const bf16* brow3 = W + (size_t)(n0 + 48 + lm) * IN_;

  v8f acc0 = {}, acc1 = {}, acc2 = {}, acc3 = {};
  for (int k0 = 0; k0 < IN_; k0 += 32) {
    v16bf a  = load_a(arow, k0, hi);
    v16bf b0 = load_b(brow0, k0, hi);
    v16bf b1 = load_b(brow1, k0, hi);
    v16bf b2 = load_b(brow2, k0, hi);
    v16bf b3 = load_b(brow3, k0, hi);
    acc0 = wmma_bf16(a, b0, acc0);
    acc1 = wmma_bf16(a, b1, acc1);
    acc2 = wmma_bf16(a, b2, acc2);
    acc3 = wmma_bf16(a, b3, acc3);
  }
  float* oplane = xp + (((size_t)dir * S_ + t) * B_) * G3_;
  v8f* accs[4] = {&acc0, &acc1, &acc2, &acc3};
#pragma unroll
  for (int j = 0; j < 4; ++j) {
    const int g = n0 + 16 * j + lm;
    const float bias = bih[g];
#pragma unroll
    for (int i = 0; i < 8; ++i)
      oplane[(size_t)(b0 + i + 8 * hi) * G3_ + g] = (*accs[j])[i] + bias;
  }
}

// ------------------------------------------------- one GRU step (both dirs)
// Block: 16 b-rows x 64 h-cols; 4 waves; each wave computes the r, z and n
// gate tiles for its 16-col sub-slab from ONE A(h) fragment per k-step.
__global__ void k_gru_step(const bf16* __restrict__ h_in,   // [2][B][H]
                           bf16* __restrict__ h_out,        // [2][B][H]
                           const bf16* __restrict__ whh,    // [2][G3*H]
                           const float* __restrict__ bhh_f,
                           const float* __restrict__ bhh_b,
                           const float* __restrict__ xp,    // [2][S][B][G3]
                           const int* __restrict__ lens,    // [B]
                           float* __restrict__ outg,        // [B][S][2H]
                           int t)
{
  __shared__ float hp[16][192];            // [b_local][ r(64) | z(64) | n(64) ]
  const int cb  = blockIdx.x;              // 0..3  : 64-col block of H
  const int bb  = blockIdx.y;              // 0..7  : 16-row block of B
  const int dir = blockIdx.z;
  const int tid = threadIdx.x;             // 0..127
  const int nsub = tid >> 5, lane = tid & 31;   // wave = 16-col sub-tile
  const int lm = lane & 15, hi = lane >> 4;

  const int b0 = bb * 16;
  const int nc = cb * 64 + nsub * 16;      // h-column base of this wave's tiles
  const bf16*  A   = h_in + (size_t)dir * B_ * H_;
  const bf16*  W   = whh + (size_t)dir * G3_ * H_;
  const float* bhh = dir ? bhh_b : bhh_f;

  const bf16* arow  = A + (size_t)(b0 + lm) * H_;
  const bf16* browR = W + (size_t)(0 * H_ + nc + lm) * H_;
  const bf16* browZ = W + (size_t)(1 * H_ + nc + lm) * H_;
  const bf16* browN = W + (size_t)(2 * H_ + nc + lm) * H_;

  v8f accR = {}, accZ = {}, accN = {};
  for (int k0 = 0; k0 < H_; k0 += 32) {
    v16bf a  = load_a(arow, k0, hi);
    v16bf bR = load_b(browR, k0, hi);
    v16bf bZ = load_b(browZ, k0, hi);
    v16bf bN = load_b(browN, k0, hi);
    accR = wmma_bf16(a, bR, accR);
    accZ = wmma_bf16(a, bZ, accZ);
    accN = wmma_bf16(a, bN, accN);
  }
  {
    v8f* accs[3] = {&accR, &accZ, &accN};
#pragma unroll
    for (int s = 0; s < 3; ++s) {
      const float bias = bhh[s * H_ + nc + lm];
      const int cl = s * 64 + nsub * 16 + lm;
#pragma unroll
      for (int i = 0; i < 8; ++i) hp[i + 8 * hi][cl] = (*accs[s])[i] + bias;
    }
  }
  __syncthreads();

  const int t_eff = dir ? (S_ - 1 - t) : t;
  const float* xplane = xp + (((size_t)dir * S_ + t) * B_) * G3_;
  for (int idx = tid; idx < 16 * 64; idx += 128) {
    int bl = idx >> 6, hc = idx & 63;
    int b  = b0 + bl;
    int gh = cb * 64 + hc;                 // h index in [0, 256)
    const float* xpb = xplane + (size_t)b * G3_;
    float xr = xpb[gh], xz = xpb[H_ + gh], xn = xpb[2 * H_ + gh];
    float hr = hp[bl][hc], hz = hp[bl][64 + hc], hn = hp[bl][128 + hc];
    float r  = sigmoidf_(xr + hr);
    float z  = sigmoidf_(xz + hz);
    float nn = tanhf(xn + r * hn);
    float hold = (float)A[(size_t)b * H_ + gh];
    float hnew = (1.0f - z) * nn + z * hold;
    h_out[((size_t)dir * B_ + b) * H_ + gh] = (bf16)hnew;
    float ov = (t_eff < lens[b]) ? hnew : 0.0f;
    outg[((size_t)b * S_ + t_eff) * (2 * H_) + dir * H_ + gh] = ov;
  }
}

// ------------------------------------------------- projection GEMM
// proj[m][c] = sum_d out[m][d] * proj_w[c][d] + proj_b[c],  m = b*S + s
__global__ void k_proj(const float* __restrict__ outg,  // [16384][512]
                       const bf16* __restrict__ pw,     // [C][512] bf16
                       const float* __restrict__ pb,
                       float* __restrict__ proj)        // [16384][C]
{
  const int nt = blockIdx.x;               // 0..3 : 64-wide c block
  const int wave = threadIdx.x >> 5, lane = threadIdx.x & 31;
  const int lm = lane & 15, hi = lane >> 4;
  const int m0 = (blockIdx.y * 8 + wave) * 16;
  const int n0 = nt * 64;

  const float* arow = outg + (size_t)(m0 + lm) * (2 * H_);
  const bf16* brow0 = pw + (size_t)(n0 +  0 + lm) * (2 * H_);
  const bf16* brow1 = pw + (size_t)(n0 + 16 + lm) * (2 * H_);
  const bf16* brow2 = pw + (size_t)(n0 + 32 + lm) * (2 * H_);
  const bf16* brow3 = pw + (size_t)(n0 + 48 + lm) * (2 * H_);

  v8f acc0 = {}, acc1 = {}, acc2 = {}, acc3 = {};
  for (int k0 = 0; k0 < 2 * H_; k0 += 32) {
    v16bf a;
#pragma unroll
    for (int i = 0; i < 8; ++i) {
      int kb = (i < 4 ? 2 * i : 2 * i + 8) + 8 * hi + k0;
      a[2 * i]     = (bf16)arow[kb];
      a[2 * i + 1] = (bf16)arow[kb + 1];
    }
    v16bf b0 = load_b(brow0, k0, hi);
    v16bf b1 = load_b(brow1, k0, hi);
    v16bf b2 = load_b(brow2, k0, hi);
    v16bf b3 = load_b(brow3, k0, hi);
    acc0 = wmma_bf16(a, b0, acc0);
    acc1 = wmma_bf16(a, b1, acc1);
    acc2 = wmma_bf16(a, b2, acc2);
    acc3 = wmma_bf16(a, b3, acc3);
  }
  v8f* accs[4] = {&acc0, &acc1, &acc2, &acc3};
#pragma unroll
  for (int j = 0; j < 4; ++j) {
    const float bias = pb[n0 + 16 * j + lm];
#pragma unroll
    for (int i = 0; i < 8; ++i)
      proj[(size_t)(m0 + i + 8 * hi) * C_ + n0 + 16 * j + lm] = (*accs[j])[i] + bias;
  }
}

// ------------------------------------------------- BN stats per sentence slot s
__global__ void k_bnstats(const float* __restrict__ proj, float* __restrict__ mu,
                          float* __restrict__ rs) {
  __shared__ float s1[256], s2[256];
  const int s = blockIdx.x, tid = threadIdx.x;
  float a = 0.0f, b = 0.0f;
  for (int i = tid; i < B_ * C_; i += 256) {
    int bb = i >> 8, c = i & 255;
    float v = proj[(((size_t)bb * S_) + s) * C_ + c];
    a += v; b += v * v;
  }
  s1[tid] = a; s2[tid] = b; __syncthreads();
  for (int o = 128; o > 0; o >>= 1) {
    if (tid < o) { s1[tid] += s1[tid + o]; s2[tid] += s2[tid + o]; }
    __syncthreads();
  }
  if (tid == 0) {
    float m = s1[0] / (float)(B_ * C_);
    float v = s2[0] / (float)(B_ * C_) - m * m;
    mu[s] = m;
    rs[s] = rsqrtf(v + 1e-5f);
  }
}

// ------------------------------------------------- attention scores
__global__ void k_scores(const float* __restrict__ proj, const float* __restrict__ mu,
                         const float* __restrict__ rs, const float* __restrict__ gamma,
                         const float* __restrict__ beta, const float* __restrict__ ctx,
                         float* __restrict__ scores) {
  __shared__ float red[256];
  const int bs = blockIdx.x;               // b*S + s
  const int sidx = bs & (S_ - 1);
  const int tid = threadIdx.x;             // channel c
  float v = (proj[(size_t)bs * C_ + tid] - mu[sidx]) * rs[sidx] * gamma[sidx] + beta[sidx];
  v = fmaxf(v, 0.0f) * ctx[tid];
  red[tid] = v; __syncthreads();
  for (int o = 128; o > 0; o >>= 1) { if (tid < o) red[tid] += red[tid + o]; __syncthreads(); }
  if (tid == 0) scores[bs] = red[0];
}

// ------------------------------------------------- softmax + weighted pooling
__global__ void k_final(const float* __restrict__ scores, const float* __restrict__ outg,
                        float* __restrict__ out) {
  __shared__ float red[256];
  __shared__ float attn[S_];
  const int b = blockIdx.x, tid = threadIdx.x;
  float sc = (tid < S_) ? scores[b * S_ + tid] : -3.4e38f;
  red[tid] = sc; __syncthreads();
  for (int o = 128; o > 0; o >>= 1) { if (tid < o) red[tid] = fmaxf(red[tid], red[tid + o]); __syncthreads(); }
  float mx = red[0]; __syncthreads();
  float e = (tid < S_) ? __expf(sc - mx) : 0.0f;
  red[tid] = e; __syncthreads();
  for (int o = 128; o > 0; o >>= 1) { if (tid < o) red[tid] += red[tid + o]; __syncthreads(); }
  float inv = 1.0f / red[0];
  if (tid < S_) attn[tid] = e * inv;
  __syncthreads();
  for (int d = tid; d < 2 * H_; d += 256) {
    float acc = 0.0f;
    for (int s = 0; s < S_; ++s)
      acc += attn[s] * outg[((size_t)b * S_ + s) * (2 * H_) + d];
    out[(size_t)b * (2 * H_) + d] = acc;
  }
}

extern "C" void kernel_launch(void* const* d_in, const int* in_sizes, int n_in,
                              void* d_out, int out_size, void* d_ws, size_t ws_size,
                              hipStream_t stream) {
  (void)in_sizes; (void)n_in; (void)out_size; (void)ws_size;
  const float* x      = (const float*)d_in[0];
  // d_in[1] (sent_hidden_stat) unused by the forward pass
  const int*   lens   = (const int*)d_in[2];
  const float* w_ih_f = (const float*)d_in[3];
  const float* w_hh_f = (const float*)d_in[4];
  const float* b_ih_f = (const float*)d_in[5];
  const float* b_hh_f = (const float*)d_in[6];
  const float* w_ih_b = (const float*)d_in[7];
  const float* w_hh_b = (const float*)d_in[8];
  const float* b_ih_b = (const float*)d_in[9];
  const float* b_hh_b = (const float*)d_in[10];
  const float* proj_w = (const float*)d_in[11];
  const float* proj_b = (const float*)d_in[12];
  const float* gamma  = (const float*)d_in[13];
  const float* beta   = (const float*)d_in[14];
  const float* ctx    = (const float*)d_in[15];

  char* ws = (char*)d_ws;
  size_t off = 0;
  float* xp   = (float*)(ws + off); off += (size_t)2 * S_ * B_ * G3_ * 4;      // 100.7 MB
  float* outg = (float*)(ws + off); off += (size_t)B_ * S_ * 2 * H_ * 4;       // 33.6 MB
  float* proj = (float*)(ws + off); off += (size_t)B_ * S_ * C_ * 4;           // 16.8 MB
  bf16*  xbf  = (bf16*)(ws + off);  off += (size_t)B_ * S_ * IN_ * 2;          // 16.8 MB
  bf16*  wih  = (bf16*)(ws + off);  off += (size_t)2 * G3_ * IN_ * 2;          // 1.5 MB
  bf16*  whh  = (bf16*)(ws + off);  off += (size_t)2 * G3_ * H_ * 2;           // 0.8 MB
  bf16*  pwbf = (bf16*)(ws + off);  off += (size_t)C_ * 2 * H_ * 2;            // 0.25 MB
  bf16*  hpp0 = (bf16*)(ws + off);  off += (size_t)2 * B_ * H_ * 2;            // 128 KB
  bf16*  hpp1 = (bf16*)(ws + off);  off += (size_t)2 * B_ * H_ * 2;            // 128 KB
  float* mu   = (float*)(ws + off); off += 256 * 4;
  float* rs   = (float*)(ws + off); off += 256 * 4;
  float* scr  = (float*)(ws + off); off += (size_t)B_ * S_ * 4;

  // --- precision conversions to bf16 ---
  k_cvt_bf16<<<256, 256, 0, stream>>>(x, xbf, B_ * S_ * IN_);
  k_cvt_bf16<<<128, 256, 0, stream>>>(w_ih_f, wih, G3_ * IN_);
  k_cvt_bf16<<<128, 256, 0, stream>>>(w_ih_b, wih + (size_t)G3_ * IN_, G3_ * IN_);
  k_cvt_bf16<<<64, 256, 0, stream>>>(w_hh_f, whh, G3_ * H_);
  k_cvt_bf16<<<64, 256, 0, stream>>>(w_hh_b, whh + (size_t)G3_ * H_, G3_ * H_);
  k_cvt_bf16<<<32, 256, 0, stream>>>(proj_w, pwbf, C_ * 2 * H_);
  (void)hipMemsetAsync(hpp0, 0, (size_t)2 * B_ * H_ * 2, stream);  // h0 = 0 (both dirs)

  // --- all input projections, both directions, one big WMMA GEMM ---
  k_input_proj<<<dim3(12, 128, 2), 256, 0, stream>>>(xbf, wih, b_ih_f, b_ih_b, xp);

  // --- sequential GRU scan: 128 fused GEMM+gate steps, h ping-pong ---
  for (int t = 0; t < S_; ++t) {
    bf16* hin  = (t & 1) ? hpp1 : hpp0;
    bf16* hout = (t & 1) ? hpp0 : hpp1;
    k_gru_step<<<dim3(4, 8, 2), 128, 0, stream>>>(hin, hout, whh, b_hh_f, b_hh_b,
                                                  xp, lens, outg, t);
  }

  // --- projection, BN stats, scores, softmax+pool ---
  k_proj<<<dim3(4, 128), 256, 0, stream>>>(outg, pwbf, proj_b, proj);
  k_bnstats<<<128, 256, 0, stream>>>(proj, mu, rs);
  k_scores<<<B_ * S_, 256, 0, stream>>>(proj, mu, rs, gamma, beta, ctx, scr);
  k_final<<<B_, 256, 0, stream>>>(scr, outg, (float*)d_out);
}
